// MultiHeadSelfAttention_42271068127379
// MI455X (gfx1250) — matrix operand order repaired and checked by reading, not compiled, not run on hardware
//
#include <hip/hip_runtime.h>

// ---------------- types ----------------
typedef __attribute__((ext_vector_type(16))) __bf16 v16bf;
typedef __attribute__((ext_vector_type(8)))  __bf16 v8bf;
typedef __attribute__((ext_vector_type(8)))  float  v8f;
typedef int v4i_vs __attribute__((vector_size(16))); // matches builtin proto

#define D_MODEL 1024
#define SEQ     2048
#define NHEAD   16
#define DK      64
#define BATCH   2
#define MTOT    (BATCH*SEQ) /* 4096 */

#define WMMA_BF16(a, b, c) \
  __builtin_amdgcn_wmma_f32_16x16x32_bf16(false, (a), false, (b), (short)0, (c), false, false)

union V16U { v16bf v; v8bf h[2]; };

__device__ __forceinline__ v16bf make_v16(v8bf lo, v8bf hi) {
  V16U u; u.h[0] = lo; u.h[1] = hi; return u.v;
}

// A-operand: 16x32 bf16 tile (M x K), row-major source, row stride ld.
// Lane l<16 holds row l, K 0..7 (elts 0..7) and K 16..23 (elts 8..15);
// lane l>=16 holds row l-16, K 8..15 and K 24..31.  (ISA 7.12.2)
__device__ __forceinline__ v16bf load_a16x32(const __bf16* base, int ld, int lane) {
  const int half = lane >> 4, r = lane & 15;
  const __bf16* p = base + r * ld + half * 8;
  v8bf lo = *(const v8bf*)(p);
  v8bf hi = *(const v8bf*)(p + 16);
  return make_v16(lo, hi);
}

// B-operand: 32x16 bf16 tile (K x N) sourced from a row-major [N, K] array
// (i.e. B^T in memory, K contiguous). Lane l<16 = column l, K 0..7 and K 16..23;
// lane l>=16 = column l-16, K 8..15 and K 24..31.
__device__ __forceinline__ v16bf load_bT16x32(const __bf16* base, int ld, int lane) {
  const int half = lane >> 4, r = lane & 15;
  const __bf16* p = base + r * ld + half * 8;
  return make_v16(*(const v8bf*)(p), *(const v8bf*)(p + 16));
}

// -------- gfx1250 async global->LDS copy (16B per lane), ASYNCcnt tracked ----
__device__ __forceinline__ void async_copy16(__bf16* lds, const __bf16* g) {
#if __has_builtin(__builtin_amdgcn_global_load_async_to_lds_b128)
  __builtin_amdgcn_global_load_async_to_lds_b128(
      (__attribute__((address_space(1))) v4i_vs*)g,
      (__attribute__((address_space(3))) v4i_vs*)lds, 0, 0);
#else
  unsigned ldsoff = (unsigned)(uintptr_t)(void*)lds; // low 32 bits = LDS offset
  asm volatile("global_load_async_to_lds_b128 %0, %1, off"
               :: "v"(ldsoff), "v"(g)
               : "memory");
#endif
}

__device__ __forceinline__ void async_wait0() {
#if __has_builtin(__builtin_amdgcn_s_wait_asynccnt)
  __builtin_amdgcn_s_wait_asynccnt(0);
#else
  asm volatile("s_wait_asynccnt 0x0" ::: "memory");
#endif
}

// ---------------- fp32 -> bf16 convert ----------------
__global__ void cvt_f32_bf16(const float* __restrict__ in, __bf16* __restrict__ out, int n) {
  int i = blockIdx.x * blockDim.x + threadIdx.x;
  int stride = gridDim.x * blockDim.x;
  for (; i < n; i += stride) out[i] = (__bf16)in[i];
}

// ---------------- GEMM: C[M,N] = A[M,K] * W[N,K]^T ----------------
// Block = 8 waves, tile 128(M) x 64(N). The 64x32 weight tile is shared by all
// 8 waves -> staged once per block into LDS via async copies (double-buffered),
// overlapping the copy of step k+1 with the WMMAs of step k.
// MODE 0: store bf16 scattered to [B, H, S, dk]   (Q, K)
// MODE 1: store bf16 scattered to [B, H, dk, S]   (V transposed)
// MODE 2: store f32 row-major [M, N]              (final Wo projection)
template <int MODE>
__global__ __launch_bounds__(256) void gemm_wmma(const __bf16* __restrict__ A,
                                                 const __bf16* __restrict__ W,
                                                 __bf16* __restrict__ outb,
                                                 float* __restrict__ outf) {
  const int K = D_MODEL;
  const int tid  = threadIdx.x;
  const int wave = tid >> 5;
  const int lane = tid & 31;
  const int half = lane >> 4, r = lane & 15;
  const int m0 = blockIdx.y * 128 + wave * 16;
  const int n0 = blockIdx.x * 64;

  __shared__ __align__(16) __bf16 bsh[2][64 * 32]; // 2 x 4KB weight tiles [n][k]

  // staging map: 256 threads x 16B = one 64x32 bf16 tile
  const int srow   = tid >> 2;        // 0..63 (weight row n0+srow)
  const int schunk = (tid & 3) * 8;   // bf16 offset within the 32-wide k slice
  const __bf16* wsrc = W + (size_t)(n0 + srow) * K + schunk;
  __bf16* sdst0 = &bsh[0][srow * 32 + schunk];
  __bf16* sdst1 = &bsh[1][srow * 32 + schunk];

  async_copy16(sdst0, wsrc); // prologue: stage k=0 into buffer 0

  v8f acc[4] = {};
  const int nsteps = K / 32;
  for (int i = 0; i < nsteps; ++i) {
    const int k = i * 32;
    async_wait0();      // this wave's staged copy landed
    __syncthreads();    // whole tile visible; previous readers drained
    if (i + 1 < nsteps)
      async_copy16((i & 1) ? sdst0 : sdst1, wsrc + k + 32);

    const __bf16* bb = &bsh[i & 1][0];
    v16bf a = load_a16x32(A + (size_t)m0 * K + k, K, lane);
    acc[0] = WMMA_BF16(a, load_bT16x32(bb +  0 * 32, 32, lane), acc[0]);
    acc[1] = WMMA_BF16(a, load_bT16x32(bb + 16 * 32, 32, lane), acc[1]);
    acc[2] = WMMA_BF16(a, load_bT16x32(bb + 32 * 32, 32, lane), acc[2]);
    acc[3] = WMMA_BF16(a, load_bT16x32(bb + 48 * 32, 32, lane), acc[3]);
    __syncthreads();    // done reading buf i&1 before it is restaged at i+2
  }

#pragma unroll
  for (int t = 0; t < 4; ++t) {
#pragma unroll
    for (int v = 0; v < 8; ++v) {
      const int row = m0 + v + 8 * half;     // C layout: M = v + 8*(lane>=16)
      const int col = n0 + t * 16 + r;       // N = lane % 16
      const float val = acc[t][v];
      if (MODE == 2) {
        outf[(size_t)row * D_MODEL + col] = val;
      } else {
        const int b = row >> 11, s = row & (SEQ - 1);
        const int h = col >> 6,  d = col & (DK - 1);
        if (MODE == 0)
          outb[(((size_t)(b * NHEAD + h)) * SEQ + s) * DK + d] = (__bf16)val;
        else
          outb[(((size_t)(b * NHEAD + h)) * DK + d) * SEQ + s] = (__bf16)val;
      }
    }
  }
}

// ---------------- causal flash attention ----------------
// One wave = one 16-row Q tile over full dk=64. 4 waves/block. Causal trip
// counts diverge per wave, so no block barriers here; LDS scratch is per-wave.
__global__ __launch_bounds__(128) void attn_wmma(const __bf16* __restrict__ Qg,
                                                 const __bf16* __restrict__ Kg,
                                                 const __bf16* __restrict__ VTg,
                                                 __bf16* __restrict__ Og) {
  const int b = blockIdx.z, h = blockIdx.y;
  const int wave = threadIdx.x >> 5, lane = threadIdx.x & 31;
  const int half = lane >> 4, r = lane & 15;
  const int q0 = blockIdx.x * 64 + wave * 16;

  const __bf16* Qbh = Qg  + (size_t)(b * NHEAD + h) * SEQ * DK; // [S, dk]
  const __bf16* Kbh = Kg  + (size_t)(b * NHEAD + h) * SEQ * DK; // [S, dk]
  const __bf16* Vbh = VTg + (size_t)(b * NHEAD + h) * DK * SEQ; // [dk, S]

  // Q tile as two A-operands (dk 0..31, dk 32..63)
  const v16bf qa0 = load_a16x32(Qbh + (size_t)q0 * DK +  0, DK, lane);
  const v16bf qa1 = load_a16x32(Qbh + (size_t)q0 * DK + 32, DK, lane);

  v8f o[4] = {};          // O accumulator, 16 x 64
  float mrow[8], lrow[8]; // per-lane row stats (row = q0 + v + 8*half)
#pragma unroll
  for (int v = 0; v < 8; ++v) { mrow[v] = -1e30f; lrow[v] = 0.0f; }

  __shared__ __align__(32) __bf16 psh[4][16 * 32];
  __bf16* pb = &psh[wave][0];

  const float scale = 0.125f; // 1/sqrt(64)
  const int kend = q0 + 16;   // causal: keys <= q0+15

  for (int k0 = 0; k0 < kend; k0 += 32) {
    // scores: S = Q * K^T, two 16x16 tiles (keys k0..+15, k0+16..+31)
    v8f s0 = {}, s1 = {};
    s0 = WMMA_BF16(qa0, load_bT16x32(Kbh + (size_t)(k0     ) * DK +  0, DK, lane), s0);
    s0 = WMMA_BF16(qa1, load_bT16x32(Kbh + (size_t)(k0     ) * DK + 32, DK, lane), s0);
    s1 = WMMA_BF16(qa0, load_bT16x32(Kbh + (size_t)(k0 + 16) * DK +  0, DK, lane), s1);
    s1 = WMMA_BF16(qa1, load_bT16x32(Kbh + (size_t)(k0 + 16) * DK + 32, DK, lane), s1);

    // mask + online softmax update, write P (bf16) to per-wave LDS scratch
#pragma unroll
    for (int v = 0; v < 8; ++v) {
      const int row = q0 + v + 8 * half;
      float a = (k0 + r      <= row) ? s0[v] * scale : -1e30f;
      float c = (k0 + 16 + r <= row) ? s1[v] * scale : -1e30f;
      float mx = fmaxf(a, c);
#pragma unroll
      for (int m = 1; m < 16; m <<= 1) mx = fmaxf(mx, __shfl_xor(mx, m, 32));
      const float newm = fmaxf(mrow[v], mx);
      const float corr = __expf(mrow[v] - newm);
      const float p0 = __expf(a - newm);
      const float p1 = __expf(c - newm);
      float rs = p0 + p1;
#pragma unroll
      for (int m = 1; m < 16; m <<= 1) rs += __shfl_xor(rs, m, 32);
      lrow[v] = lrow[v] * corr + rs;
      mrow[v] = newm;
#pragma unroll
      for (int t = 0; t < 4; ++t) o[t][v] *= corr;
      const int rr = v + 8 * half;
      pb[rr * 32 + r]      = (__bf16)p0;
      pb[rr * 32 + 16 + r] = (__bf16)p1;
    }

    // P (16x32) back out of LDS in A-operand layout (wave-private; dscnt waits
    // inserted by compiler), then O += P * V using transposed V (keys contiguous)
    const v16bf pa = load_a16x32(pb, 32, lane);
#pragma unroll
    for (int t = 0; t < 4; ++t) {
      v16bf vb = load_bT16x32(Vbh + (size_t)(t * 16) * SEQ + k0, SEQ, lane);
      o[t] = WMMA_BF16(pa, vb, o[t]);
    }
  }

  // normalize and store to attn-out [B, S, D] bf16 (gathered head layout)
#pragma unroll
  for (int t = 0; t < 4; ++t) {
#pragma unroll
    for (int v = 0; v < 8; ++v) {
      const int row = q0 + v + 8 * half;
      const int col = h * DK + t * 16 + r;
      Og[(size_t)(b * SEQ + row) * D_MODEL + col] = (__bf16)(o[t][v] / lrow[v]);
    }
  }
}

// ---------------- launch ----------------
extern "C" void kernel_launch(void* const* d_in, const int* in_sizes, int n_in,
                              void* d_out, int out_size, void* d_ws, size_t ws_size,
                              hipStream_t stream) {
  const float* x  = (const float*)d_in[0];
  const float* Wq = (const float*)d_in[1];
  const float* Wk = (const float*)d_in[2];
  const float* Wv = (const float*)d_in[3];
  const float* Wo = (const float*)d_in[4];
  float* out = (float*)d_out;

  char* ws = (char*)d_ws;
  const size_t MB = 1ull << 20;
  __bf16* xb  = (__bf16*)(ws +  0 * MB); // 8 MiB: x bf16 [4096,1024]
  __bf16* wqb = (__bf16*)(ws +  8 * MB); // 2 MiB each
  __bf16* wkb = (__bf16*)(ws + 10 * MB);
  __bf16* wvb = (__bf16*)(ws + 12 * MB);
  __bf16* wob = (__bf16*)(ws + 14 * MB);
  __bf16* Qb  = (__bf16*)(ws + 16 * MB); // 8 MiB [B,H,S,dk]
  __bf16* Kb  = (__bf16*)(ws + 24 * MB); // 8 MiB [B,H,S,dk]
  __bf16* VTb = (__bf16*)(ws + 32 * MB); // 8 MiB [B,H,dk,S]
  __bf16* Ab  = (__bf16*)(ws + 40 * MB); // 8 MiB attn out bf16 [B,S,D]

  const int nx = MTOT * D_MODEL;      // 4M
  const int nw = D_MODEL * D_MODEL;   // 1M
  cvt_f32_bf16<<<2048, 256, 0, stream>>>(x,  xb,  nx);
  cvt_f32_bf16<<<1024, 256, 0, stream>>>(Wq, wqb, nw);
  cvt_f32_bf16<<<1024, 256, 0, stream>>>(Wk, wkb, nw);
  cvt_f32_bf16<<<1024, 256, 0, stream>>>(Wv, wvb, nw);
  cvt_f32_bf16<<<1024, 256, 0, stream>>>(Wo, wob, nw);

  dim3 gg(D_MODEL / 64, MTOT / 128);  // (16, 32), 8 waves/block
  gemm_wmma<0><<<gg, 256, 0, stream>>>(xb, wqb, Qb,  nullptr);
  gemm_wmma<0><<<gg, 256, 0, stream>>>(xb, wkb, Kb,  nullptr);
  gemm_wmma<1><<<gg, 256, 0, stream>>>(xb, wvb, VTb, nullptr);

  dim3 ga(SEQ / 64, NHEAD, BATCH);    // (32, 16, 2), 4 waves/block
  attn_wmma<<<ga, 128, 0, stream>>>(Qb, Kb, VTb, Ab);

  gemm_wmma<2><<<gg, 256, 0, stream>>>(Ab, wob, nullptr, out);
}
